// QLoRABigNet_39221641347363
// MI455X (gfx1250) — matrix operand order repaired and checked by reading, not compiled, not run on hardware
//
#include <hip/hip_runtime.h>
#include <stdint.h>

#define DD    1024
#define BATCH 16384
#define GSZ   16
#define NGRP  65536
#define RANK  32
#define LN_EPS 1e-5f

typedef __attribute__((ext_vector_type(16))) __bf16 bf16x16;
typedef __attribute__((ext_vector_type(8)))  float  f32x8;

union Frag { uint4 q[2]; bf16x16 v; };

// ---- CDNA5 async global->LDS path (ASYNCcnt) via inline asm: portable across
// toolchains whose __builtin_amdgcn_global_load_async_to_lds_* signatures vary.
#if defined(__HIP_DEVICE_COMPILE__) && defined(__gfx1250__)
#define USE_ASYNC 1
#else
#define USE_ASYNC 0
#endif

#if USE_ASYNC
__device__ __forceinline__ void async_cp16(const void* g, void* l) {
  unsigned           lds32 = (unsigned)(uintptr_t)l;        // low 32 bits give the LDS address
  unsigned long long ga    = (unsigned long long)(uintptr_t)g;
  asm volatile("global_load_async_to_lds_b128 %0, %1, off"
               :: "v"(lds32), "v"(ga) : "memory");
}
#define ASYNC_WAIT() asm volatile("s_wait_asynccnt 0x0" ::: "memory")
#define KBUF 2
#else
#define KBUF 1
#endif

__device__ __forceinline__ uint16_t f2bf(float f) {
  unsigned u = __float_as_uint(f);
  u += 0x7FFFu + ((u >> 16) & 1u);          // round-to-nearest-even
  return (uint16_t)(u >> 16);
}
__device__ __forceinline__ float bf2f(uint16_t h) {
  return __uint_as_float(((unsigned)h) << 16);
}

// ---------------------------------------------------------------- fp32 -> bf16
__global__ __launch_bounds__(256) void k_cvt_bf16(const float* __restrict__ in,
                                                  uint16_t* __restrict__ out, int n4) {
  int i = blockIdx.x * 256 + threadIdx.x;
  if (i >= n4) return;
  float4 v = reinterpret_cast<const float4*>(in)[i];
  reinterpret_cast<ushort4*>(out)[i] =
      make_ushort4(f2bf(v.x), f2bf(v.y), f2bf(v.z), f2bf(v.w));
}

// ------------------------------------------------------- int4-group -> bf16 W
__global__ __launch_bounds__(256) void k_dequant(const int* __restrict__ q,
                                                 const float* __restrict__ nrm,
                                                 uint16_t* __restrict__ wout) {
  int g = blockIdx.x * 256 + threadIdx.x;     // 65536 groups per layer
  float n = nrm[g];
  const int* qg = q + (size_t)g * GSZ;
  union { uint16_t h[16]; uint4 q4[2]; } o;
#pragma unroll
  for (int j = 0; j < 16; ++j)
    o.h[j] = f2bf(((float)qg[j] * (2.0f / 15.0f) - 1.0f) * n);
  uint4* dst = reinterpret_cast<uint4*>(wout + (size_t)g * GSZ);
  dst[0] = o.q4[0];
  dst[1] = o.q4[1];
}

// ------------------------------------------------- t[B,32] = Hbf @ LAbf[32,D]^T
__global__ __launch_bounds__(256) void k_lora_t(const uint16_t* __restrict__ H,
                                                const uint16_t* __restrict__ LA,
                                                uint16_t* __restrict__ T) {
  __shared__ uint16_t sA[128 * 72];
  const int tid  = threadIdx.x;
  const int lane = tid & 31, w = tid >> 5;
  const int lr   = lane & 15, hi = lane >> 4;
  const int m0   = blockIdx.x * 128;
  f32x8 acc[2] = {};
  for (int k0 = 0; k0 < DD; k0 += 64) {
    __syncthreads();
    for (int c = tid; c < 128 * 8; c += 256) {
      int row = c >> 3, col = (c & 7) * 8;
      *reinterpret_cast<uint4*>(&sA[row * 72 + col]) =
          *reinterpret_cast<const uint4*>(H + (size_t)(m0 + row) * DD + k0 + col);
    }
    __syncthreads();
#pragma unroll
    for (int ks = 0; ks < 64; ks += 32) {
      Frag a;
      int row = w * 16 + lr;
      int kb  = ks + hi * 8;
      a.q[0] = *reinterpret_cast<const uint4*>(&sA[row * 72 + kb]);
      a.q[1] = *reinterpret_cast<const uint4*>(&sA[row * 72 + kb + 16]);
#pragma unroll
      for (int j = 0; j < 2; ++j) {
        Frag b;
        int n  = j * 16 + lr;
        int kg = k0 + ks + hi * 8;
        b.q[0] = *reinterpret_cast<const uint4*>(LA + (size_t)n * DD + kg);
        b.q[1] = *reinterpret_cast<const uint4*>(LA + (size_t)n * DD + kg + 16);
        acc[j] = __builtin_amdgcn_wmma_f32_16x16x32_bf16(
            false, a.v, false, b.v, (short)0, acc[j], false, false);
      }
    }
  }
#pragma unroll
  for (int j = 0; j < 2; ++j) {
    int n = j * 16 + lr;
#pragma unroll
    for (int vg = 0; vg < 8; ++vg) {
      int m = m0 + w * 16 + vg + hi * 8;
      T[(size_t)m * RANK + n] = f2bf(acc[j][vg]);
    }
  }
}

// --------------------------------------------- out = H@W^T (+ t@LB^T) + bias
// 128x128 tile per block, 8 wave32s, each wave 32x64 (2x4 WMMA 16x16x32 bf16)
__global__ __launch_bounds__(256) void k_gemm(const uint16_t* __restrict__ H,
                                              const uint16_t* __restrict__ W,
                                              const float* __restrict__ bias,
                                              const uint16_t* __restrict__ T,
                                              const uint16_t* __restrict__ LB,
                                              uint16_t* __restrict__ O,
                                              int relu) {
  __shared__ uint16_t sA[KBUF][128 * 72];
  __shared__ uint16_t sB[KBUF][128 * 72];
  const int tid  = threadIdx.x;
  const int lane = tid & 31, w = tid >> 5;
  const int lr   = lane & 15, hi = lane >> 4;
  const int wm   = (w & 3) * 32;   // 4 waves along M
  const int wn   = (w >> 2) * 64;  // 2 waves along N
  const int n0   = blockIdx.x * 128;
  const int m0   = blockIdx.y * 128;
  f32x8 acc[2][4] = {};

  auto mma_slab = [&](const uint16_t* cA, const uint16_t* cB) {
#pragma unroll
    for (int ks = 0; ks < 64; ks += 32) {
      Frag a[2], b[4];
#pragma unroll
      for (int i = 0; i < 2; ++i) {
        int row = wm + i * 16 + lr;
        int kb  = ks + hi * 8;
        a[i].q[0] = *reinterpret_cast<const uint4*>(&cA[row * 72 + kb]);
        a[i].q[1] = *reinterpret_cast<const uint4*>(&cA[row * 72 + kb + 16]);
      }
#pragma unroll
      for (int j = 0; j < 4; ++j) {
        int n  = wn + j * 16 + lr;
        int kb = ks + hi * 8;
        b[j].q[0] = *reinterpret_cast<const uint4*>(&cB[n * 72 + kb]);
        b[j].q[1] = *reinterpret_cast<const uint4*>(&cB[n * 72 + kb + 16]);
      }
#pragma unroll
      for (int i = 0; i < 2; ++i)
#pragma unroll
        for (int j = 0; j < 4; ++j)
          acc[i][j] = __builtin_amdgcn_wmma_f32_16x16x32_bf16(
              false, a[i].v, false, b[j].v, (short)0, acc[i][j], false, false);
    }
  };

#if USE_ASYNC
  // double-buffered: ASYNCcnt-tracked global->LDS DMA overlapped with WMMA
  auto stage_async = [&](int k0, int buf) {
    for (int c = tid; c < 128 * 8; c += 256) {
      int row = c >> 3, col = (c & 7) * 8;
      async_cp16(H + (size_t)(m0 + row) * DD + k0 + col, &sA[buf][row * 72 + col]);
      async_cp16(W + (size_t)(n0 + row) * DD + k0 + col, &sB[buf][row * 72 + col]);
    }
  };
  stage_async(0, 0);
  for (int kt = 0; kt < DD / 64; ++kt) {
    ASYNC_WAIT();          // my async writes into buf[kt&1] have landed
    __syncthreads();       // everyone's have landed; prior reads of buf done
    if (kt + 1 < DD / 64) stage_async((kt + 1) * 64, (kt + 1) & 1);
    mma_slab(sA[kt & 1], sB[kt & 1]);
  }
#else
  for (int k0 = 0; k0 < DD; k0 += 64) {
    __syncthreads();
    for (int c = tid; c < 128 * 8; c += 256) {
      int row = c >> 3, col = (c & 7) * 8;
      *reinterpret_cast<uint4*>(&sA[0][row * 72 + col]) =
          *reinterpret_cast<const uint4*>(H + (size_t)(m0 + row) * DD + k0 + col);
      *reinterpret_cast<uint4*>(&sB[0][row * 72 + col]) =
          *reinterpret_cast<const uint4*>(W + (size_t)(n0 + row) * DD + k0 + col);
      if (k0 + 64 < DD && (c & 7) == 0) {   // warm next K-slab
        __builtin_prefetch(H + (size_t)(m0 + row) * DD + k0 + 64, 0, 0);
        __builtin_prefetch(W + (size_t)(n0 + row) * DD + k0 + 64, 0, 0);
      }
    }
    __syncthreads();
    mma_slab(sA[0], sB[0]);
  }
#endif

  // LoRA as K-extension: out += t @ LB^T  (K = 32, one WMMA round)
  if (T != nullptr) {
    __syncthreads();
#if USE_ASYNC
    for (int c = tid; c < 128 * 4; c += 256) {
      int row = c >> 2, col = (c & 3) * 8;
      async_cp16(T + (size_t)(m0 + row) * RANK + col, &sA[0][row * 72 + col]);
      async_cp16(LB + (size_t)(n0 + row) * RANK + col, &sB[0][row * 72 + col]);
    }
    ASYNC_WAIT();
    __syncthreads();
#else
    for (int c = tid; c < 128 * 4; c += 256) {
      int row = c >> 2, col = (c & 3) * 8;
      *reinterpret_cast<uint4*>(&sA[0][row * 72 + col]) =
          *reinterpret_cast<const uint4*>(T + (size_t)(m0 + row) * RANK + col);
      *reinterpret_cast<uint4*>(&sB[0][row * 72 + col]) =
          *reinterpret_cast<const uint4*>(LB + (size_t)(n0 + row) * RANK + col);
    }
    __syncthreads();
#endif
    Frag a[2], b[4];
#pragma unroll
    for (int i = 0; i < 2; ++i) {
      int row = wm + i * 16 + lr;
      int kb  = hi * 8;
      a[i].q[0] = *reinterpret_cast<const uint4*>(&sA[0][row * 72 + kb]);
      a[i].q[1] = *reinterpret_cast<const uint4*>(&sA[0][row * 72 + kb + 16]);
    }
#pragma unroll
    for (int j = 0; j < 4; ++j) {
      int n  = wn + j * 16 + lr;
      int kb = hi * 8;
      b[j].q[0] = *reinterpret_cast<const uint4*>(&sB[0][n * 72 + kb]);
      b[j].q[1] = *reinterpret_cast<const uint4*>(&sB[0][n * 72 + kb + 16]);
    }
#pragma unroll
    for (int i = 0; i < 2; ++i)
#pragma unroll
      for (int j = 0; j < 4; ++j)
        acc[i][j] = __builtin_amdgcn_wmma_f32_16x16x32_bf16(
            false, a[i].v, false, b[j].v, (short)0, acc[i][j], false, false);
  }

  // epilogue: +bias, optional ReLU, store bf16
#pragma unroll
  for (int i = 0; i < 2; ++i)
#pragma unroll
    for (int j = 0; j < 4; ++j) {
      int   n  = n0 + wn + j * 16 + lr;
      float bs = bias[n];
#pragma unroll
      for (int vg = 0; vg < 8; ++vg) {
        int   m = m0 + wm + i * 16 + vg + hi * 8;
        float v = acc[i][j][vg] + bs;
        if (relu) v = fmaxf(v, 0.0f);
        O[(size_t)m * DD + n] = f2bf(v);
      }
    }
}

// -------------------------------------------------------------- LayerNorm row
__global__ __launch_bounds__(256) void k_ln(const uint16_t* __restrict__ H,
                                            const float* __restrict__ lnw,
                                            const float* __restrict__ lnb,
                                            uint16_t* __restrict__ Hout,
                                            float* __restrict__ Fout) {
  __shared__ float red1[256];
  __shared__ float red2[256];
  int row = blockIdx.x, tid = threadIdx.x;
  float v[4];
  float s1 = 0.0f, s2 = 0.0f;
#pragma unroll
  for (int i = 0; i < 4; ++i) {
    v[i] = bf2f(H[(size_t)row * DD + tid * 4 + i]);
    s1 += v[i];
    s2 += v[i] * v[i];
  }
  red1[tid] = s1; red2[tid] = s2;
  __syncthreads();
  for (int off = 128; off > 0; off >>= 1) {
    if (tid < off) { red1[tid] += red1[tid + off]; red2[tid] += red2[tid + off]; }
    __syncthreads();
  }
  float mu  = red1[0] * (1.0f / 1024.0f);
  float var = red2[0] * (1.0f / 1024.0f) - mu * mu;
  float rs  = rsqrtf(var + LN_EPS);
#pragma unroll
  for (int i = 0; i < 4; ++i) {
    int   c = tid * 4 + i;
    float y = (v[i] - mu) * rs * lnw[c] + lnb[c];
    if (Fout) Fout[(size_t)row * DD + c] = y;
    else      Hout[(size_t)row * DD + c] = f2bf(y);
  }
}

// ---------------------------------------------------------------------- host
extern "C" void kernel_launch(void* const* d_in, const int* in_sizes, int n_in,
                              void* d_out, int out_size, void* d_ws, size_t ws_size,
                              hipStream_t stream) {
  (void)in_sizes; (void)n_in; (void)out_size; (void)ws_size;
  const float* x    = (const float*)d_in[0];
  const int*   wq   = (const int*)d_in[1];
  const float* wn   = (const float*)d_in[2];
  const float* bias = (const float*)d_in[3];
  const float* la   = (const float*)d_in[4];
  const float* lb   = (const float*)d_in[5];
  const float* lnw  = (const float*)d_in[6];
  const float* lnb  = (const float*)d_in[7];
  float* out = (float*)d_out;

  char*  ws  = (char*)d_ws;
  size_t off = 0;
  auto alloc = [&](size_t bytes) -> void* {
    off = (off + 255) & ~(size_t)255;
    void* p = ws + off;
    off += bytes;
    return p;
  };
  uint16_t* hA   = (uint16_t*)alloc((size_t)BATCH * DD * 2);
  uint16_t* hB   = (uint16_t*)alloc((size_t)BATCH * DD * 2);
  uint16_t* Wbf  = (uint16_t*)alloc((size_t)DD * DD * 2);
  uint16_t* Tbf  = (uint16_t*)alloc((size_t)BATCH * RANK * 2);
  uint16_t* LAbf = (uint16_t*)alloc((size_t)3 * RANK * DD * 2);
  uint16_t* LBbf = (uint16_t*)alloc((size_t)3 * DD * RANK * 2);

  { int n4 = BATCH * DD / 4;  k_cvt_bf16<<<(n4 + 255) / 256, 256, 0, stream>>>(x,  hA,   n4); }
  { int n4 = 3 * RANK * DD/4; k_cvt_bf16<<<(n4 + 255) / 256, 256, 0, stream>>>(la, LAbf, n4); }
  { int n4 = 3 * DD * RANK/4; k_cvt_bf16<<<(n4 + 255) / 256, 256, 0, stream>>>(lb, LBbf, n4); }

  uint16_t* cur = hA;
  uint16_t* nxt = hB;
  for (int l = 0; l < 18; ++l) {
    int blk = l / 3, pos = l % 3;
    k_dequant<<<NGRP / 256, 256, 0, stream>>>(wq + (size_t)l * NGRP * GSZ,
                                              wn + (size_t)l * NGRP, Wbf);
    int slot = (l == 0) ? 0 : (l == 6) ? 1 : (l == 12) ? 2 : -1;
    const uint16_t* Tptr  = nullptr;
    const uint16_t* LBptr = nullptr;
    if (slot >= 0) {
      k_lora_t<<<BATCH / 128, 256, 0, stream>>>(cur, LAbf + (size_t)slot * RANK * DD, Tbf);
      Tptr  = Tbf;
      LBptr = LBbf + (size_t)slot * DD * RANK;
    }
    k_gemm<<<dim3(DD / 128, BATCH / 128), 256, 0, stream>>>(
        cur, Wbf, bias + (size_t)l * DD, Tptr, LBptr, nxt, pos < 2 ? 1 : 0);
    uint16_t* t = cur; cur = nxt; nxt = t;
    if (pos == 2) {
      bool fin = (blk == 5);
      k_ln<<<BATCH, 256, 0, stream>>>(cur, lnw + (size_t)blk * DD,
                                      lnb + (size_t)blk * DD, cur,
                                      fin ? out : nullptr);
    }
  }
}
